// ASFormerBlock_69758858822513
// MI455X (gfx1250) — matrix operand order repaired and checked by reading, not compiled, not run on hardware
//
#include <hip/hip_runtime.h>
#include <hip/hip_bf16.h>

#define D_    512
#define H_    8
#define DH_   64
#define T_    1024
#define B_    4
#define DFF_  2048
#define MAXREL_ 256
#define PSTR  528   // 33*16, padded width of rel-bias P slice

typedef __attribute__((ext_vector_type(16))) __bf16 v16bf;
typedef __attribute__((ext_vector_type(8)))  float  v8f;

union Frag {
  v16bf v;
  unsigned int u[8];
};

static __device__ inline unsigned short f2bf(float f) {
  unsigned int u = __float_as_uint(f);
  u += 0x7FFFu + ((u >> 16) & 1u);      // round-to-nearest-even
  return (unsigned short)(u >> 16);
}
static __device__ inline float bf2f(unsigned short h) {
  return __uint_as_float(((unsigned int)h) << 16);
}
static __device__ inline unsigned int pack2f(float a, float b) {
  return (unsigned int)f2bf(a) | ((unsigned int)f2bf(b) << 16);
}
static __device__ inline v8f wmma_bf16(const Frag& a, const Frag& b, v8f c) {
  return __builtin_amdgcn_wmma_f32_16x16x32_bf16(false, a.v, false, b.v,
                                                 (short)0, c, false, false);
}
// A-matrix 16-bit layout (ISA 7.12.2): VGPR v holds K pair
static __device__ inline int kA(int v, int hh) {
  return (v < 4) ? (v * 2 + hh * 8) : (16 + (v - 4) * 2 + hh * 8);
}
// B-matrix 16-bit layout: same K order as A (pairs per VGPR)
static __device__ inline int kB(int v, int hh) { return kA(v, hh); }

// CDNA5 async global->LDS copy, 16 bytes per lane (ASYNCcnt-tracked).
static __device__ inline void async16(unsigned lds_off, const unsigned short* src) {
  unsigned long long ga = (unsigned long long)(size_t)src;
  asm volatile("global_load_async_to_lds_b128 %0, %1, off"
               :: "v"(lds_off), "v"(ga) : "memory");
}
#define WAIT_ASYNC3() asm volatile("s_wait_asynccnt 0x3" ::: "memory")
#define WAIT_ASYNC0() asm volatile("s_wait_asynccnt 0x0" ::: "memory")

// ------------- one-time repacks: weights -> transposed bf16 (N,K) --------
__global__ void zero_page(unsigned short* zp) {
  if (threadIdx.x < 512) zp[threadIdx.x] = 0;
}
__global__ void transpose_bf16(const float* __restrict__ w,
                               unsigned short* __restrict__ wt, int K, int N) {
  int idx = blockIdx.x * blockDim.x + threadIdx.x;   // over N*K
  if (idx >= N * K) return;
  int n = idx / K, k = idx % K;
  wt[idx] = f2bf(w[(size_t)k * N + n]);
}
// conv_w (O,I,K=3) -> w2t[o][tap*D+ci]  (already (N,Ktot) transposed)
__global__ void repack_conv_bf16(const float* __restrict__ cw,
                                 unsigned short* __restrict__ w2t) {
  int idx = blockIdx.x * blockDim.x + threadIdx.x;   // over D * 3D
  if (idx >= D_ * 3 * D_) return;
  int o = idx / (3 * D_), k = idx % (3 * D_);
  int tap = k / D_, ci = k % D_;
  w2t[idx] = f2bf(cw[((size_t)o * D_ + ci) * 3 + tap]);
}

// ---------------- LayerNorm: f32 in -> bf16 out --------------------------
__global__ __launch_bounds__(256) void ln_kernel(const float* __restrict__ x,
                                                 const float* __restrict__ w,
                                                 const float* __restrict__ b,
                                                 unsigned short* __restrict__ o) {
  __shared__ float rbuf[8];
  const int row = blockIdx.x;
  const int tid = threadIdx.x;
  const int lane = tid & 31, wv = tid >> 5;
  const float* xr = x + (size_t)row * D_;
  float v0 = xr[tid], v1 = xr[tid + 256];
  float s = v0 + v1;
  for (int off = 16; off; off >>= 1) s += __shfl_xor(s, off, 32);
  if (lane == 0) rbuf[wv] = s;
  __syncthreads();
  float tot = 0.f;
  for (int i = 0; i < 8; ++i) tot += rbuf[i];
  float mu = tot * (1.0f / D_);
  __syncthreads();
  float d0 = v0 - mu, d1 = v1 - mu;
  float q = d0 * d0 + d1 * d1;
  for (int off = 16; off; off >>= 1) q += __shfl_xor(q, off, 32);
  if (lane == 0) rbuf[wv] = q;
  __syncthreads();
  float vtot = 0.f;
  for (int i = 0; i < 8; ++i) vtot += rbuf[i];
  float inv = rsqrtf(vtot * (1.0f / D_) + 1e-5f);
  unsigned short* orow = o + (size_t)row * D_;
  orow[tid]       = f2bf(d0 * inv * w[tid]       + b[tid]);
  orow[tid + 256] = f2bf(d1 * inv * w[tid + 256] + b[tid + 256]);
}

// ---------------- async double-buffered bf16 WMMA GEMM -------------------
// C[M,N] = act(A[M,Ktot] @ Bt[N,Ktot]^T + bias) (+ residual, f32 out only)
// A, Bt are bf16; tiles staged via GLOBAL_LOAD_ASYNC_TO_LDS_B128.
#define TM 128
#define TN 64
__global__ __launch_bounds__(256) void gemm_bf16(
    const unsigned short* __restrict__ A, const unsigned short* __restrict__ Bt,
    const float* __restrict__ bias, const float* __restrict__ residual,
    void* __restrict__ Cout, int M, int N, int Ktot,
    int act, int outBf16, int convMode, const unsigned short* __restrict__ zp) {
  __shared__ unsigned short lsa[2][TM][32];
  __shared__ unsigned short lsb[2][TN][32];   // [n][k] layout
  const int tid  = threadIdx.x;
  const int lane = tid & 31;
  const int wv   = tid >> 5;
  const int wm = wv & 3, wn = wv >> 2;        // 4x2 wave grid, 32x32 per wave
  const int m0 = blockIdx.x * TM;
  const int n0 = blockIdx.y * TN;
  const int hh = lane >> 4, l16 = lane & 15;

  v8f acc[2][2];
  for (int mi = 0; mi < 2; ++mi)
    for (int ni = 0; ni < 2; ++ni)
      for (int v = 0; v < 8; ++v) acc[mi][ni][v] = 0.0f;

  // issue async copies for the K-chunk starting at k0 into buffer bufI:
  // exactly 3 async instructions per wave (2 for A tile, 1 for B tile).
  auto issue_tile = [&](int k0, int bufI) {
    for (int idx = tid; idx < TM * 4; idx += 256) {
      int r = idx >> 2, seg = idx & 3;
      int row = m0 + r;
      int kg = k0 + seg * 8;
      const unsigned short* src;
      if (!convMode) {
        src = A + (size_t)row * Ktot + kg;
      } else {  // im2col of h (B_,T_,D_), KSIZE=3, pad=1, per-batch bounds
        int bb = row >> 10, t = row & (T_ - 1);
        int tap = kg >> 9, ci = kg & (D_ - 1);
        int ts = t + tap - 1;
        src = (ts >= 0 && ts < T_) ? A + (((size_t)(bb * T_ + ts)) << 9) + ci
                                   : zp;
      }
      async16((unsigned)(size_t)&lsa[bufI][r][seg * 8], src);
    }
    for (int idx = tid; idx < TN * 4; idx += 256) {
      int n = idx >> 2, seg = idx & 3;
      async16((unsigned)(size_t)&lsb[bufI][n][seg * 8],
              Bt + (size_t)(n0 + n) * Ktot + k0 + seg * 8);
    }
  };

  const int nk = Ktot / 32;
  issue_tile(0, 0);
  for (int kt = 0; kt < nk; ++kt) {
    const int cur = kt & 1;
    if (kt + 1 < nk) {
      issue_tile((kt + 1) * 32, cur ^ 1);
      WAIT_ASYNC3();               // 3 newest (next tile) may stay in flight
    } else {
      WAIT_ASYNC0();
    }
    __syncthreads();               // tile `cur` visible to all waves

    Frag af[2], bfr[2];
    for (int mi = 0; mi < 2; ++mi) {
      int row = wm * 32 + mi * 16 + l16;
      for (int v = 0; v < 8; ++v)
        af[mi].u[v] = *(const unsigned int*)&lsa[cur][row][kA(v, hh)];
    }
    for (int ni = 0; ni < 2; ++ni) {
      int col = wn * 32 + ni * 16 + l16;
      for (int v = 0; v < 8; ++v)
        bfr[ni].u[v] = *(const unsigned int*)&lsb[cur][col][kB(v, hh)];
    }
    for (int mi = 0; mi < 2; ++mi)
      for (int ni = 0; ni < 2; ++ni)
        acc[mi][ni] = wmma_bf16(af[mi], bfr[ni], acc[mi][ni]);
    __syncthreads();               // done reading `cur` before it is rewritten
  }

  for (int mi = 0; mi < 2; ++mi)
    for (int ni = 0; ni < 2; ++ni) {
      int col = n0 + wn * 32 + ni * 16 + l16;
      float bb = bias ? bias[col] : 0.0f;
      for (int v = 0; v < 8; ++v) {
        int row = m0 + wm * 32 + mi * 16 + v + hh * 8;
        float x = acc[mi][ni][v] + bb;
        if (act == 1)      x = fmaxf(x, 0.0f);
        else if (act == 2) x = 0.5f * x * (1.0f + erff(x * 0.70710678118654752f));
        if (outBf16) {
          ((unsigned short*)Cout)[(size_t)row * N + col] = f2bf(x);
        } else {
          if (residual) x += residual[(size_t)row * N + col];
          ((float*)Cout)[(size_t)row * N + col] = x;
        }
      }
    }
}

// ---------------- flash attention with relative positional bias ----------
// bf16 Q/K/V in, bf16 attn-out. grid (T/32, H, B), block 64 (2 waves).
__global__ __launch_bounds__(64) void attn_kernel(
    const unsigned short* __restrict__ Q, const unsigned short* __restrict__ K,
    const unsigned short* __restrict__ V, const float* __restrict__ rel_table,
    const unsigned char* __restrict__ mask, unsigned short* __restrict__ O) {
  __shared__ unsigned short lp[2][16][PSTR];  // per-wave scaled bias P slice
  __shared__ unsigned short ls[2][16][32];    // per-wave prob staging (C->A)
  const int lane = threadIdx.x & 31;
  const int wv   = threadIdx.x >> 5;
  const int hh = lane >> 4, l16 = lane & 15;
  const int b  = blockIdx.z, hd = blockIdx.y;
  const int i0 = (blockIdx.x * 2 + wv) * 16;

  Frag qa[2];
  {
    const unsigned short* Qp = Q + ((size_t)(b * T_ + i0 + l16)) * D_ + hd * DH_;
    for (int kc = 0; kc < 2; ++kc)
      for (int v = 0; v < 8; ++v)
        qa[kc].u[v] = *(const unsigned int*)(Qp + kA(v, hh) + kc * 32);
  }

  // Phase 1: P = (Q @ rel_table^T) / sqrt(DH) -> LDS bf16.
  for (int rt = 0; rt < 33; ++rt) {
    int r = rt * 16 + l16;
    Frag rb[2];
    for (int kc = 0; kc < 2; ++kc)
      for (int v = 0; v < 8; ++v) {
        int d = kB(v, hh) + kc * 32;
        float a0 = 0.f, a1 = 0.f;
        if (r < 2 * MAXREL_ + 1) {
          const float* p = rel_table + (size_t)r * DH_ + d;
          a0 = p[0]; a1 = p[1];
        }
        rb[kc].u[v] = pack2f(a0, a1);
      }
    v8f pc;
    for (int v = 0; v < 8; ++v) pc[v] = 0.0f;
    pc = wmma_bf16(qa[0], rb[0], pc);
    pc = wmma_bf16(qa[1], rb[1], pc);
    for (int v = 0; v < 8; ++v)
      lp[wv][v + hh * 8][rt * 16 + l16] = f2bf(pc[v] * 0.125f);
  }

  v8f oacc[4];
  for (int t = 0; t < 4; ++t)
    for (int v = 0; v < 8; ++v) oacc[t][v] = 0.0f;
  float mrow[8], lrow[8];
  for (int v = 0; v < 8; ++v) { mrow[v] = -INFINITY; lrow[v] = 0.0f; }

  for (int jt = 0; jt < T_ / 32; ++jt) {
    const int j0 = jt * 32;
    Frag kb[2][2];
    for (int ni = 0; ni < 2; ++ni) {
      const unsigned short* Kp =
          K + ((size_t)(b * T_ + j0 + ni * 16 + l16)) * D_ + hd * DH_;
      for (int kc = 0; kc < 2; ++kc)
        for (int v = 0; v < 8; ++v)
          kb[ni][kc].u[v] = *(const unsigned int*)(Kp + kB(v, hh) + kc * 32);
    }
    v8f s[2];
    for (int ni = 0; ni < 2; ++ni) {
      for (int v = 0; v < 8; ++v) s[ni][v] = 0.0f;
      s[ni] = wmma_bf16(qa[0], kb[ni][0], s[ni]);
      s[ni] = wmma_bf16(qa[1], kb[ni][1], s[ni]);
    }
    float tmax[8];
    for (int v = 0; v < 8; ++v) tmax[v] = -INFINITY;
    for (int ni = 0; ni < 2; ++ni) {
      int j = j0 + ni * 16 + l16;
      bool mk = mask ? (mask[b * T_ + j] != 0) : true;
      for (int v = 0; v < 8; ++v) {
        int irow = v + hh * 8;
        int d = j - (i0 + irow);
        d = d < -MAXREL_ ? -MAXREL_ : (d > MAXREL_ ? MAXREL_ : d);
        float sc = s[ni][v] * 0.125f + bf2f(lp[wv][irow][d + MAXREL_]);
        sc = mk ? sc : -INFINITY;
        s[ni][v] = sc;
        tmax[v] = fmaxf(tmax[v], sc);
      }
    }
    for (int off = 1; off < 16; off <<= 1)
      for (int v = 0; v < 8; ++v)
        tmax[v] = fmaxf(tmax[v], __shfl_xor(tmax[v], off, 32));
    float rsum[8];
    for (int v = 0; v < 8; ++v) {
      float mnew = fmaxf(mrow[v], tmax[v]);
      float scl = (mrow[v] == -INFINITY) ? 0.0f : expf(mrow[v] - mnew);
      lrow[v] *= scl;
      for (int t = 0; t < 4; ++t) oacc[t][v] *= scl;
      mrow[v] = mnew;
      rsum[v] = 0.0f;
    }
    for (int ni = 0; ni < 2; ++ni)
      for (int v = 0; v < 8; ++v) {
        float p = (s[ni][v] == -INFINITY) ? 0.0f : expf(s[ni][v] - mrow[v]);
        rsum[v] += p;
        ls[wv][v + hh * 8][ni * 16 + l16] = f2bf(p);
      }
    for (int off = 1; off < 16; off <<= 1)
      for (int v = 0; v < 8; ++v) rsum[v] += __shfl_xor(rsum[v], off, 32);
    for (int v = 0; v < 8; ++v) lrow[v] += rsum[v];

    Frag sf;
    for (int v = 0; v < 8; ++v)
      sf.u[v] = *(const unsigned int*)&ls[wv][l16][kA(v, hh)];
    for (int t = 0; t < 4; ++t) {
      Frag vb;
      int dh = t * 16 + l16;
      for (int v = 0; v < 8; ++v) {
        int key = kB(v, hh);
        const unsigned short* Vp =
            V + ((size_t)(b * T_ + j0 + key)) * D_ + hd * DH_ + dh;
        vb.u[v] = (unsigned int)Vp[0] | ((unsigned int)Vp[D_] << 16);
      }
      oacc[t] = wmma_bf16(sf, vb, oacc[t]);
    }
  }

  for (int v = 0; v < 8; ++v) {
    float inv = (lrow[v] > 0.0f) ? 1.0f / lrow[v] : 0.0f;
    int row = i0 + v + hh * 8;
    for (int t = 0; t < 4; ++t)
      O[((size_t)(b * T_ + row)) * D_ + hd * DH_ + t * 16 + l16] =
          f2bf(oacc[t][v] * inv);
  }
}

// ---------------- host-side orchestration -------------------------------
extern "C" void kernel_launch(void* const* d_in, const int* in_sizes, int n_in,
                              void* d_out, int out_size, void* d_ws, size_t ws_size,
                              hipStream_t stream) {
  const float* x      = (const float*)d_in[0];
  const unsigned char* mask = (const unsigned char*)d_in[1];
  const float* ln1_w  = (const float*)d_in[2];
  const float* ln1_b  = (const float*)d_in[3];
  const float* conv_w = (const float*)d_in[4];
  const float* conv_b = (const float*)d_in[5];
  const float* ln2_w  = (const float*)d_in[6];
  const float* ln2_b  = (const float*)d_in[7];
  const float* q_w    = (const float*)d_in[8];
  const float* q_b    = (const float*)d_in[9];
  const float* k_w    = (const float*)d_in[10];
  const float* k_b    = (const float*)d_in[11];
  const float* v_w    = (const float*)d_in[12];
  const float* v_b    = (const float*)d_in[13];
  const float* o_w    = (const float*)d_in[14];
  const float* o_b    = (const float*)d_in[15];
  const float* rel_t  = (const float*)d_in[16];
  const float* ln3_w  = (const float*)d_in[17];
  const float* ln3_b  = (const float*)d_in[18];
  const float* fc1_w  = (const float*)d_in[19];
  const float* fc1_b  = (const float*)d_in[20];
  const float* fc2_w  = (const float*)d_in[21];
  const float* fc2_b  = (const float*)d_in[22];
  float* out = (float*)d_out;

  const int M = B_ * T_;                        // 4096
  unsigned char* base = (unsigned char*)d_ws;
  size_t off = 0;
  auto carve = [&](size_t bytes) {
    void* p = base + off;
    off += (bytes + 255) & ~(size_t)255;
    return p;
  };
  unsigned short* W2t  = (unsigned short*)carve((size_t)D_ * 3 * D_ * 2);
  unsigned short* qwT  = (unsigned short*)carve((size_t)D_ * D_ * 2);
  unsigned short* kwT  = (unsigned short*)carve((size_t)D_ * D_ * 2);
  unsigned short* vwT  = (unsigned short*)carve((size_t)D_ * D_ * 2);
  unsigned short* owT  = (unsigned short*)carve((size_t)D_ * D_ * 2);
  unsigned short* f1T  = (unsigned short*)carve((size_t)DFF_ * D_ * 2);
  unsigned short* f2T  = (unsigned short*)carve((size_t)D_ * DFF_ * 2);
  unsigned short* h    = (unsigned short*)carve((size_t)M * D_ * 2);
  unsigned short* Qb   = (unsigned short*)carve((size_t)M * D_ * 2);
  unsigned short* Kb   = (unsigned short*)carve((size_t)M * D_ * 2);
  unsigned short* Vb   = (unsigned short*)carve((size_t)M * D_ * 2);
  unsigned short* g    = (unsigned short*)carve((size_t)M * DFF_ * 2);
  float*          x1   = (float*)carve((size_t)M * D_ * 4);
  float*          x2   = (float*)carve((size_t)M * D_ * 4);
  unsigned short* zp   = (unsigned short*)carve(512 * 2);
  unsigned short* attnout = h;                  // h dead after QKV gemms
  unsigned short* h3      = Qb;                 // Q dead after attention

  dim3 blk256(256), blk64(64);
  dim3 gLN(M);
  dim3 gG512(M / TM, D_ / TN);                  // (32, 8)
  dim3 gG2048(M / TM, DFF_ / TN);               // (32, 32)
  dim3 gAttn(T_ / 32, H_, B_);

  // one-time repacks (run every call; deterministic)
  zero_page<<<1, 512, 0, stream>>>(zp);
  repack_conv_bf16<<<(D_ * 3 * D_ + 255) / 256, blk256, 0, stream>>>(conv_w, W2t);
  transpose_bf16<<<(D_ * D_ + 255) / 256, blk256, 0, stream>>>(q_w, qwT, D_, D_);
  transpose_bf16<<<(D_ * D_ + 255) / 256, blk256, 0, stream>>>(k_w, kwT, D_, D_);
  transpose_bf16<<<(D_ * D_ + 255) / 256, blk256, 0, stream>>>(v_w, vwT, D_, D_);
  transpose_bf16<<<(D_ * D_ + 255) / 256, blk256, 0, stream>>>(o_w, owT, D_, D_);
  transpose_bf16<<<(D_ * DFF_ + 255) / 256, blk256, 0, stream>>>(fc1_w, f1T, D_, DFF_);
  transpose_bf16<<<(D_ * DFF_ + 255) / 256, blk256, 0, stream>>>(fc2_w, f2T, DFF_, D_);

  // sublayer 1: x1 = x + relu(conv(LN1(x)))
  ln_kernel<<<gLN, blk256, 0, stream>>>(x, ln1_w, ln1_b, h);
  gemm_bf16<<<gG512, blk256, 0, stream>>>(h, W2t, conv_b, x, x1,
                                          M, D_, 3 * D_, /*relu*/1, 0,
                                          /*conv*/1, zp);
  // sublayer 2: attention
  ln_kernel<<<gLN, blk256, 0, stream>>>(x1, ln2_w, ln2_b, h);
  gemm_bf16<<<gG512, blk256, 0, stream>>>(h, qwT, q_b, nullptr, Qb,
                                          M, D_, D_, 0, /*bf16*/1, 0, zp);
  gemm_bf16<<<gG512, blk256, 0, stream>>>(h, kwT, k_b, nullptr, Kb,
                                          M, D_, D_, 0, 1, 0, zp);
  gemm_bf16<<<gG512, blk256, 0, stream>>>(h, vwT, v_b, nullptr, Vb,
                                          M, D_, D_, 0, 1, 0, zp);
  attn_kernel<<<gAttn, blk64, 0, stream>>>(Qb, Kb, Vb, rel_t, mask, attnout);
  gemm_bf16<<<gG512, blk256, 0, stream>>>(attnout, owT, o_b, x1, x2,
                                          M, D_, D_, 0, 0, 0, zp);
  // sublayer 3: FFN
  ln_kernel<<<gLN, blk256, 0, stream>>>(x2, ln3_w, ln3_b, h3);
  gemm_bf16<<<gG2048, blk256, 0, stream>>>(h3, f1T, fc1_b, nullptr, g,
                                           M, DFF_, D_, /*gelu*/2, 1, 0, zp);
  gemm_bf16<<<gG512, blk256, 0, stream>>>(g, f2T, fc2_b, x2, out,
                                          M, D_, DFF_, 0, 0, 0, zp);
}
